// GraphVAE_70987219468420
// MI455X (gfx1250) — matrix operand order repaired and checked by reading, not compiled, not run on hardware
//
#include <hip/hip_runtime.h>
#include <hip/hip_fp16.h>

typedef __attribute__((ext_vector_type(16))) _Float16 v16h;
typedef __attribute__((ext_vector_type(8)))  _Float16 v8h;
typedef __attribute__((ext_vector_type(8)))  float    v8f;

#define NNODES  65536
#define NEDGES  1048576
#define NGRAPHS 512
#define FIN     7
#define HDIM    128
#define ZDIM    64
#define MAXN    28

// ---------------------------------------------------------------------------
// CDNA5 async global->LDS helpers (ASYNCcnt-tracked, VFLAT encoding)
// ---------------------------------------------------------------------------
typedef __attribute__((address_space(3))) char lds_char;

__device__ __forceinline__ unsigned lds_off(const void* p) {
  // generic -> LDS addrspace cast strips the aperture, leaving the byte offset
  return (unsigned)(size_t)(lds_char*)(p);
}

__device__ __forceinline__ void async_ld_b128(unsigned ldsOff, const void* g) {
  asm volatile("global_load_async_to_lds_b128 %0, %1, off"
               :: "v"(ldsOff), "v"((unsigned long long)(size_t)g)
               : "memory");
}

__device__ __forceinline__ void wait_async0() {
  asm volatile("s_wait_asynccnt 0x0" ::: "memory");
}

// ---------------------------------------------------------------------------
// WMMA GEMM: C[M x N](f16) = [relu](A[M x 128](f16) * B[128 x N] + bias) [+ res]
// B supplied pre-transposed as f16 Bt[N][128].
// grid.x = M/128, grid.y = N/128, block = 256 (8 waves).
// Both 32KB tiles are contiguous in global -> staged with async b128 copies.
// ---------------------------------------------------------------------------
__global__ __launch_bounds__(256) void k_gemm128(
    const _Float16* __restrict__ A, const _Float16* __restrict__ Bt,
    const float* __restrict__ bias, const _Float16* __restrict__ res,
    _Float16* __restrict__ C, int N, int doRelu)
{
  __shared__ _Float16 Alds[128][128];   // [m][k]
  __shared__ _Float16 Blds[128][128];   // [n][k]

  const int t = threadIdx.x;
  const int mBase = blockIdx.x * 128;
  const int nBase = blockIdx.y * 128;

  // Async-stage both tiles (each 128*128*2B = 32KB, contiguous)
  {
    const char* Ag = (const char*)(A + ((size_t)mBase << 7));
    const char* Bg = (const char*)(Bt + ((size_t)nBase << 7));
    unsigned aoff = lds_off(&Alds[0][0]);
    unsigned boff = lds_off(&Blds[0][0]);
#pragma unroll
    for (int i = 0; i < 8; ++i) {
      const int e = (t + i * 256) * 16;     // byte offset, 16B per b128
      async_ld_b128(aoff + e, Ag + e);
      async_ld_b128(boff + e, Bg + e);
    }
  }
  wait_async0();
  __syncthreads();

  const int wave = t >> 5;
  const int lane = t & 31;
  const int half = lane >> 4;
  const int l15  = lane & 15;
  const int mrow = (wave << 4) + l15;

  v8f acc[8];
#pragma unroll
  for (int nt = 0; nt < 8; ++nt)
#pragma unroll
    for (int v = 0; v < 8; ++v) acc[nt][v] = 0.0f;

#pragma unroll
  for (int k0 = 0; k0 < 128; k0 += 32) {
    // A fragment 16x32 (ISA 7.12.2): lane holds K=k0+half*8+[0..7], k0+16+half*8+[0..7]
    v8h alo = *(const v8h*)&Alds[mrow][k0 + half * 8];
    v8h ahi = *(const v8h*)&Alds[mrow][k0 + 16 + half * 8];
    v16h a;
#pragma unroll
    for (int i = 0; i < 8; ++i) { a[i] = alo[i]; a[8 + i] = ahi[i]; }

#pragma unroll
    for (int nt = 0; nt < 8; ++nt) {
      const int ncol = (nt << 4) + l15;
      v8h blo = *(const v8h*)&Blds[ncol][k0 + half * 8];
      v8h bhi = *(const v8h*)&Blds[ncol][k0 + 16 + half * 8];
      v16h b;
#pragma unroll
      for (int i = 0; i < 8; ++i) { b[i] = blo[i]; b[8 + i] = bhi[i]; }

      acc[nt] = __builtin_amdgcn_wmma_f32_16x16x32_f16(
          false, a, false, b, (short)0, acc[nt], false, false);
    }
  }

  const int crow0 = (wave << 4) + half * 8;
#pragma unroll
  for (int nt = 0; nt < 8; ++nt) {
    const int n  = nBase + (nt << 4) + l15;
    const float bv = bias ? bias[n] : 0.0f;
#pragma unroll
    for (int v = 0; v < 8; ++v) {
      const size_t m = (size_t)(mBase + crow0 + v);
      float val = acc[nt][v] + bv;
      if (doRelu) val = fmaxf(val, 0.0f);
      if (res) val += (float)res[m * (size_t)N + n];
      C[m * (size_t)N + n] = (_Float16)val;
    }
  }
}

// Weight prep: W[128 x N] f32 row-major -> Bt[N][128] f16
__global__ void k_prep_bt(const float* __restrict__ W, _Float16* __restrict__ Bt, int N)
{
  int idx = blockIdx.x * blockDim.x + threadIdx.x;   // N*128
  if (idx >= N * HDIM) return;
  int n = idx >> 7, k = idx & 127;
  Bt[idx] = (_Float16)W[k * N + n];
}

// h = relu(x @ Win + bin0)   (K=7, scalar), f16 output
__global__ void k_encin(const float* __restrict__ x, const float* __restrict__ Win,
                        const float* __restrict__ bin0, _Float16* __restrict__ h)
{
  int idx = blockIdx.x * blockDim.x + threadIdx.x;   // NNODES*128
  int n = idx >> 7, j = idx & 127;
  float s = bin0[j];
#pragma unroll
  for (int f = 0; f < FIN; ++f) s += x[n * FIN + f] * Win[f * HDIM + j];
  h[idx] = (_Float16)fmaxf(s, 0.0f);
}

__global__ void k_zero32(unsigned* __restrict__ p, int n)
{
  int i = blockIdx.x * blockDim.x + threadIdx.x;
  if (i < n) p[i] = 0u;
}

// agg[dst[e]] += msg[src[e]] : one wave per edge, packed f16 atomics
// (global_atomic_pk_add_f16: 2 halves per op, 64 atomics/edge vs 128 f32)
__global__ void k_scatter(const int* __restrict__ src, const int* __restrict__ dst,
                          const _Float16* __restrict__ msg, _Float16* __restrict__ agg)
{
  int gid  = blockIdx.x * blockDim.x + threadIdx.x;
  int edge = gid >> 5;
  int lane = gid & 31;
  if (edge >= NEDGES) return;
  int s = src[edge], d = dst[edge];
  const __half2* m = (const __half2*)(msg + ((size_t)s << 7)) + lane * 2;
  __half2*       a = (__half2*)(agg + ((size_t)d << 7)) + lane * 2;
  __half2 v0 = m[0];
  __half2 v1 = m[1];
  atomicAdd(a + 0, v0);
  atomicAdd(a + 1, v1);
}

// g[batch[n]] += h[n]  (f32 accumulation for the pooled sums)
__global__ void k_readout(const _Float16* __restrict__ h, const int* __restrict__ batch,
                          float* __restrict__ g)
{
  int idx = blockIdx.x * blockDim.x + threadIdx.x;   // NNODES*128
  int n = idx >> 7, j = idx & 127;
  atomicAdd(&g[batch[n] * HDIM + j], (float)h[idx]);
}

// z_mean / z_logvar / reparameterized z (all f32)
__global__ void k_latent(const float* __restrict__ g, const float* __restrict__ Wmean,
                         const float* __restrict__ bmean, const float* __restrict__ Wlv,
                         const float* __restrict__ blv, const float* __restrict__ eps,
                         float* __restrict__ zm, float* __restrict__ zlv,
                         float* __restrict__ z)
{
  int idx = blockIdx.x * blockDim.x + threadIdx.x;   // NGRAPHS*ZDIM
  int gi = idx >> 6, j = idx & 63;
  const float* row = g + gi * HDIM;
  float m = bmean[j], lv = blv[j];
#pragma unroll 4
  for (int k = 0; k < HDIM; ++k) {
    float hv = row[k];
    m  += hv * Wmean[k * ZDIM + j];
    lv += hv * Wlv[k * ZDIM + j];
  }
  zm[idx]  = m;
  zlv[idx] = lv;
  z[idx]   = m + eps[idx] * __expf(0.5f * lv);
}

// gr1 = relu(z @ Wg1 + bg1)   (K=64), f16 output feeding WMMA
__global__ void k_dec1(const float* __restrict__ z, const float* __restrict__ Wg1,
                       const float* __restrict__ bg1, _Float16* __restrict__ gr1)
{
  int idx = blockIdx.x * blockDim.x + threadIdx.x;   // NGRAPHS*HDIM
  int gi = idx >> 7, j = idx & 127;
  float s = bg1[j];
#pragma unroll 4
  for (int k = 0; k < ZDIM; ++k) s += z[gi * ZDIM + k] * Wg1[k * HDIM + j];
  gr1[idx] = (_Float16)fmaxf(s, 0.0f);
}

// node_features = emb @ Wnf + bnf   (N=7), f32 output
__global__ void k_nodefeat(const _Float16* __restrict__ emb, const float* __restrict__ Wnf,
                           const float* __restrict__ bnf, float* __restrict__ nf)
{
  int idx = blockIdx.x * blockDim.x + threadIdx.x;   // NGRAPHS*MAXN*FIN
  int row = idx / FIN, f = idx % FIN;
  const _Float16* e = emb + (size_t)row * HDIM;
  float s = bnf[f];
#pragma unroll 4
  for (int k = 0; k < HDIM; ++k) s += (float)e[k] * Wnf[k * FIN + f];
  nf[idx] = s;
}

// Pairwise edge logits: one block per graph; hi/hj tiles in LDS (f32)
__global__ __launch_bounds__(256) void k_edges(
    const _Float16* __restrict__ hi, const _Float16* __restrict__ hj,
    const float* __restrict__ be1, const float* __restrict__ We2,
    const float* __restrict__ be2, float* __restrict__ el)
{
  __shared__ float shi[MAXN][HDIM];
  __shared__ float shj[MAXN][HDIM];
  __shared__ float sw[HDIM];
  __shared__ float sb[HDIM];
  __shared__ float raw[MAXN][MAXN];

  const int g = blockIdx.x, t = threadIdx.x;
  const size_t base = (size_t)g * MAXN * HDIM;
  for (int i = t; i < MAXN * HDIM; i += 256) {
    shi[i >> 7][i & 127] = (float)hi[base + i];
    shj[i >> 7][i & 127] = (float)hj[base + i];
  }
  if (t < HDIM) { sw[t] = We2[t]; sb[t] = be1[t]; }
  __syncthreads();

  const float b2 = be2[0];
  for (int p = t; p < MAXN * MAXN; p += 256) {
    int i = p / MAXN, j = p % MAXN;
    float s = b2;
#pragma unroll 4
    for (int k = 0; k < HDIM; ++k)
      s += fmaxf(shi[i][k] + shj[j][k] + sb[k], 0.0f) * sw[k];
    raw[i][j] = s;
  }
  __syncthreads();

  for (int p = t; p < MAXN * MAXN; p += 256) {
    int i = p / MAXN, j = p % MAXN;
    float v = (i == j) ? -1.0e9f : 0.5f * (raw[i][j] + raw[j][i]);
    el[(size_t)g * (MAXN * MAXN) + p] = v;
  }
}

// ---------------------------------------------------------------------------
extern "C" void kernel_launch(void* const* d_in, const int* in_sizes, int n_in,
                              void* d_out, int out_size, void* d_ws, size_t ws_size,
                              hipStream_t stream)
{
  const float* x     = (const float*)d_in[0];
  const int*   ei    = (const int*)d_in[1];     // [2, NEDGES]
  const int*   batch = (const int*)d_in[2];
  const float* eps   = (const float*)d_in[3];
  const float* Win   = (const float*)d_in[4];
  const float* bin0  = (const float*)d_in[5];
  const float* Wm    = (const float*)d_in[6];   // [3,128,128]
  const float* bm    = (const float*)d_in[7];
  const float* Wu    = (const float*)d_in[8];
  const float* bu    = (const float*)d_in[9];
  const float* Wmean = (const float*)d_in[10];
  const float* bmean = (const float*)d_in[11];
  const float* Wlv   = (const float*)d_in[12];
  const float* blv   = (const float*)d_in[13];
  const float* Wg1   = (const float*)d_in[14];
  const float* bg1   = (const float*)d_in[15];
  const float* Wg2   = (const float*)d_in[16];
  const float* bg2   = (const float*)d_in[17];
  const float* Wne   = (const float*)d_in[18];
  const float* bne   = (const float*)d_in[19];
  const float* Wnf   = (const float*)d_in[20];
  const float* bnf   = (const float*)d_in[21];
  const float* We1i  = (const float*)d_in[22];
  const float* We1j  = (const float*)d_in[23];
  const float* be1   = (const float*)d_in[24];
  const float* We2   = (const float*)d_in[25];
  const float* be2   = (const float*)d_in[26];

  float* out     = (float*)d_out;
  float* nf_out  = out;                              // 512*28*7   = 100352
  float* el_out  = out + 100352;                     // 512*28*28  = 401408
  float* zm_out  = out + 100352 + 401408;            // 512*64
  float* zlv_out = out + 100352 + 401408 + 32768;    // 512*64

  // Workspace layout (256B-aligned chunks)
  char* wsb = (char*)d_ws;
  auto alloc = [&](size_t bytes) { char* p = wsb; wsb += (bytes + 255) & ~(size_t)255; return p; };
  _Float16* h    = (_Float16*)alloc((size_t)NNODES * HDIM * 2);
  _Float16* msg  = (_Float16*)alloc((size_t)NNODES * HDIM * 2);
  _Float16* agg  = (_Float16*)alloc((size_t)NNODES * HDIM * 2);
  float*    g    = (float*)   alloc((size_t)NGRAPHS * HDIM * 4);
  float*    z    = (float*)   alloc((size_t)NGRAPHS * ZDIM * 4);
  _Float16* gr1  = (_Float16*)alloc((size_t)NGRAPHS * HDIM * 2);
  _Float16* gr   = (_Float16*)alloc((size_t)NGRAPHS * HDIM * 2);
  _Float16* emb  = (_Float16*)alloc((size_t)NGRAPHS * MAXN * HDIM * 2);
  _Float16* hi_  = (_Float16*)alloc((size_t)NGRAPHS * MAXN * HDIM * 2);
  _Float16* hj_  = (_Float16*)alloc((size_t)NGRAPHS * MAXN * HDIM * 2);
  _Float16* btWm = (_Float16*)alloc((size_t)3 * HDIM * HDIM * 2);
  _Float16* btWu = (_Float16*)alloc((size_t)3 * HDIM * HDIM * 2);
  _Float16* btWg2= (_Float16*)alloc((size_t)HDIM * HDIM * 2);
  _Float16* btWne= (_Float16*)alloc((size_t)MAXN * HDIM * HDIM * 2);
  _Float16* btE1i= (_Float16*)alloc((size_t)HDIM * HDIM * 2);
  _Float16* btE1j= (_Float16*)alloc((size_t)HDIM * HDIM * 2);

  // --- weight prep: f32 [128 x N] -> f16 transposed [N][128] ---
  for (int l = 0; l < 3; ++l) {
    k_prep_bt<<<(HDIM * HDIM) / 256, 256, 0, stream>>>(Wm + l * HDIM * HDIM,
                                                       btWm + l * HDIM * HDIM, HDIM);
    k_prep_bt<<<(HDIM * HDIM) / 256, 256, 0, stream>>>(Wu + l * HDIM * HDIM,
                                                       btWu + l * HDIM * HDIM, HDIM);
  }
  k_prep_bt<<<(HDIM * HDIM) / 256, 256, 0, stream>>>(Wg2, btWg2, HDIM);
  k_prep_bt<<<(MAXN * HDIM * HDIM) / 256, 256, 0, stream>>>(Wne, btWne, MAXN * HDIM);
  k_prep_bt<<<(HDIM * HDIM) / 256, 256, 0, stream>>>(We1i, btE1i, HDIM);
  k_prep_bt<<<(HDIM * HDIM) / 256, 256, 0, stream>>>(We1j, btE1j, HDIM);

  // --- encoder input MLP ---
  k_encin<<<(NNODES * HDIM) / 256, 256, 0, stream>>>(x, Win, bin0, h);

  // --- message passing layers ---
  const dim3 gNode(NNODES / 128, 1);
  for (int l = 0; l < 3; ++l) {
    k_gemm128<<<gNode, 256, 0, stream>>>(h, btWm + l * HDIM * HDIM, bm + l * HDIM,
                                         nullptr, msg, HDIM, 1);
    k_zero32<<<(NNODES * HDIM / 2) / 256, 256, 0, stream>>>((unsigned*)agg,
                                                            NNODES * HDIM / 2);
    k_scatter<<<(NEDGES * 32) / 256, 256, 0, stream>>>(ei, ei + NEDGES, msg, agg);
    // h = relu(agg @ Wu + bu) + h (elementwise in-place residual is safe)
    k_gemm128<<<gNode, 256, 0, stream>>>(agg, btWu + l * HDIM * HDIM, bu + l * HDIM,
                                         h, h, HDIM, 1);
  }

  // --- readout + latent ---
  k_zero32<<<(NGRAPHS * HDIM + 255) / 256, 256, 0, stream>>>((unsigned*)g, NGRAPHS * HDIM);
  k_readout<<<(NNODES * HDIM) / 256, 256, 0, stream>>>(h, batch, g);
  k_latent<<<(NGRAPHS * ZDIM) / 256, 256, 0, stream>>>(g, Wmean, bmean, Wlv, blv,
                                                       eps, zm_out, zlv_out, z);

  // --- decoder ---
  k_dec1<<<(NGRAPHS * HDIM) / 256, 256, 0, stream>>>(z, Wg1, bg1, gr1);
  k_gemm128<<<dim3(NGRAPHS / 128, 1), 256, 0, stream>>>(gr1, btWg2, bg2, nullptr,
                                                        gr, HDIM, 1);
  k_gemm128<<<dim3(NGRAPHS / 128, (MAXN * HDIM) / 128), 256, 0, stream>>>(
      gr, btWne, bne, nullptr, emb, MAXN * HDIM, 1);
  k_nodefeat<<<(NGRAPHS * MAXN * FIN) / 256, 256, 0, stream>>>(emb, Wnf, bnf, nf_out);
  k_gemm128<<<dim3((NGRAPHS * MAXN) / 128, 1), 256, 0, stream>>>(emb, btE1i, nullptr,
                                                                 nullptr, hi_, HDIM, 0);
  k_gemm128<<<dim3((NGRAPHS * MAXN) / 128, 1), 256, 0, stream>>>(emb, btE1j, nullptr,
                                                                 nullptr, hj_, HDIM, 0);
  k_edges<<<NGRAPHS, 256, 0, stream>>>(hi_, hj_, be1, We2, be2, el_out);
}
